// BertSelfAttentionWithExtras_9053791060219
// MI455X (gfx1250) — matrix operand order repaired and checked by reading, not compiled, not run on hardware
//
#include <hip/hip_runtime.h>

#define B_N   16
#define S_N   512
#define H_N   16
#define DH_N  64
#define D_N   1024

typedef __attribute__((ext_vector_type(16))) __bf16          v16bf;
typedef __attribute__((ext_vector_type(16))) unsigned short  v16u;
typedef __attribute__((ext_vector_type(8)))  unsigned short  v8u;
typedef __attribute__((ext_vector_type(8)))  float           v8f;
typedef __attribute__((ext_vector_type(4)))  float           v4f;
typedef __attribute__((ext_vector_type(4)))  int             v4i;

// D = A(16x32 bf16) * B(32x16 bf16) + C(16x16 f32)
#define WMMA_B16(A, B, C) __builtin_amdgcn_wmma_f32_16x16x32_bf16( \
    false, __builtin_bit_cast(v16bf, (A)),                         \
    false, __builtin_bit_cast(v16bf, (B)),                         \
    (short)0, (C), false, false)

__device__ __forceinline__ unsigned short f2bf(float f) {
  union { float f; unsigned u; } a; a.f = f;
  unsigned r = a.u + 0x7FFFu + ((a.u >> 16) & 1u);   // round-to-nearest-even
  return (unsigned short)(r >> 16);
}

// ---- CDNA5 async global->LDS copy (ASYNCcnt path), with safe fallback ------
#if __has_builtin(__builtin_amdgcn_global_load_async_to_lds_b128)
#define HAS_ASYNC_LDS 1
#else
#define HAS_ASYNC_LDS 0
#endif

__device__ __forceinline__ void async_copy16(unsigned short* lds,
                                             const unsigned short* g) {
#if HAS_ASYNC_LDS
  __builtin_amdgcn_global_load_async_to_lds_b128(
      (__attribute__((address_space(1))) v4i*)g,
      (__attribute__((address_space(3))) v4i*)lds,
      0, 0);
#else
  *(v8u*)lds = *(const v8u*)g;
#endif
}

__device__ __forceinline__ void async_wait() {
#if HAS_ASYNC_LDS
#if __has_builtin(__builtin_amdgcn_s_wait_asynccnt)
  __builtin_amdgcn_s_wait_asynccnt(0);
#else
  asm volatile("s_wait_asynccnt 0x0" ::: "memory");
#endif
#endif
}

// ---------------------------------------------------------------------------
// Kernel 1: dist_emb (1023x64 f32) -> bf16, padded to 1024 rows (row 1023 = 0)
// ---------------------------------------------------------------------------
__global__ __launch_bounds__(256)
void conv_pe_kernel(const float* __restrict__ E, unsigned short* __restrict__ Ebf) {
  const int i = blockIdx.x * 256 + threadIdx.x;
  if (i < 1024 * DH_N)
    Ebf[i] = (i < 1023 * DH_N) ? f2bf(E[i]) : (unsigned short)0;
}

// ---------------------------------------------------------------------------
// Kernel 2: fused QKV projection GEMM.  Y = X[8192x1024] @ W[1024x1024] + b
// grid.z selects (Wq,bq,q) / (Wk,bk,k) / (Wv,bv,v).  Output bf16 in [B,H,S,DH].
// Block: 256 thr = 8 waves; tile 128x128; k-step 32 (one bf16 WMMA per step).
// ---------------------------------------------------------------------------
__global__ __launch_bounds__(256)
void qkv_gemm_kernel(const float* __restrict__ X,
                     const float* __restrict__ Wq, const float* __restrict__ bq,
                     const float* __restrict__ Wk, const float* __restrict__ bk,
                     const float* __restrict__ Wv, const float* __restrict__ bv,
                     unsigned short* __restrict__ qbf,
                     unsigned short* __restrict__ kbf,
                     unsigned short* __restrict__ vbf)
{
  __shared__ unsigned short sA[128][40];   // X tile  [m][k]  bf16, padded
  __shared__ unsigned short sB[128][40];   // W tile  [n][k]  bf16 (transposed)

  const float* W; const float* bias; unsigned short* out;
  if (blockIdx.z == 0)      { W = Wq; bias = bq; out = qbf; }
  else if (blockIdx.z == 1) { W = Wk; bias = bk; out = kbf; }
  else                      { W = Wv; bias = bv; out = vbf; }

  const int t    = threadIdx.x;
  const int lane = t & 31;
  const int wave = t >> 5;
  const int half = lane >> 4;
  const int ln   = lane & 15;
  const int wm   = wave >> 1;          // 0..3 -> 32-row strip
  const int wn   = wave & 1;           // 0..1 -> 64-col strip
  const int bm0  = blockIdx.y * 128;
  const int bn0  = blockIdx.x * 128;

  const v8f vzero = {0.f,0.f,0.f,0.f,0.f,0.f,0.f,0.f};
  v8f acc[2][4];
  #pragma unroll
  for (int mi = 0; mi < 2; ++mi)
    #pragma unroll
    for (int ni = 0; ni < 4; ++ni) acc[mi][ni] = vzero;

  for (int k0 = 0; k0 < D_N; k0 += 32) {
    // ---- stage tiles (fp32 global -> bf16 LDS) ----
    {
      const int r  = t >> 3;
      const int c4 = (t & 7) * 4;
      #pragma unroll
      for (int rr = 0; rr < 128; rr += 32) {
        v4f x = *(const v4f*)(X + (size_t)(bm0 + r + rr) * D_N + k0 + c4);
        sA[r + rr][c4 + 0] = f2bf(x[0]);
        sA[r + rr][c4 + 1] = f2bf(x[1]);
        sA[r + rr][c4 + 2] = f2bf(x[2]);
        sA[r + rr][c4 + 3] = f2bf(x[3]);
      }
      const int kr = t >> 5;
      const int n4 = (t & 31) * 4;
      #pragma unroll
      for (int kk = 0; kk < 32; kk += 8) {
        v4f w = *(const v4f*)(W + (size_t)(k0 + kr + kk) * D_N + bn0 + n4);
        sB[n4 + 0][kr + kk] = f2bf(w[0]);
        sB[n4 + 1][kr + kk] = f2bf(w[1]);
        sB[n4 + 2][kr + kk] = f2bf(w[2]);
        sB[n4 + 3][kr + kk] = f2bf(w[3]);
      }
      // prefetch next k-step's tiles (global_prefetch_b8)
      if (k0 + 32 < D_N) {
        __builtin_prefetch(X + (size_t)(bm0 + r) * D_N + k0 + 32 + c4, 0, 3);
        __builtin_prefetch(W + (size_t)(k0 + 32 + kr) * D_N + bn0 + n4, 0, 3);
      }
    }
    __syncthreads();

    // ---- fragments (contiguous k = half*16 + s in A and B) ----
    v16u af[2], bfr[4];
    #pragma unroll
    for (int mi = 0; mi < 2; ++mi) {
      const int row = wm * 32 + mi * 16 + ln;            // A: lane%16 = M
      #pragma unroll
      for (int v = 0; v < 8; ++v) {
        const int kk = half * 16 + 2 * v;
        af[mi][2 * v]     = sA[row][kk];
        af[mi][2 * v + 1] = sA[row][kk + 1];
      }
    }
    #pragma unroll
    for (int ni = 0; ni < 4; ++ni) {
      const int col = wn * 64 + ni * 16 + ln;            // B: lane%16 = N, K = half*16+i
      v8u lo = *(const v8u*)&sB[col][half * 16];
      v8u hi = *(const v8u*)&sB[col][half * 16 + 8];
      #pragma unroll
      for (int i = 0; i < 8; ++i) { bfr[ni][i] = lo[i]; bfr[ni][i + 8] = hi[i]; }
    }
    #pragma unroll
    for (int mi = 0; mi < 2; ++mi)
      #pragma unroll
      for (int ni = 0; ni < 4; ++ni)
        acc[mi][ni] = WMMA_B16(af[mi], bfr[ni], acc[mi][ni]);
    __syncthreads();
  }

  // ---- bias + scatter to [B,H,S,DH] bf16 ----
  #pragma unroll
  for (int mi = 0; mi < 2; ++mi)
    #pragma unroll
    for (int ni = 0; ni < 4; ++ni) {
      const int gn = bn0 + wn * 64 + ni * 16 + ln;
      const float bia = bias[gn];
      const int hh = gn >> 6, dd = gn & 63;
      #pragma unroll
      for (int v = 0; v < 8; ++v) {                      // C: M = v + 8*half
        const int gm = bm0 + wm * 32 + mi * 16 + half * 8 + v;
        const int bb = gm >> 9, ss = gm & 511;
        out[((size_t)(bb * H_N + hh) * S_N + ss) * DH_N + dd] =
            f2bf(acc[mi][ni][v] + bia);
      }
    }
}

// ---------------------------------------------------------------------------
// Kernel 3: attention for one (b, h, 32 q-rows).  64 threads = 2 waves,
// each wave owns 16 q-rows.  Relative-position einsums become E-window GEMMs.
// q/k/v tile staging uses GLOBAL_LOAD_ASYNC_TO_LDS_B128 (ASYNCcnt).
// ---------------------------------------------------------------------------
__global__ __launch_bounds__(64)
void attn_kernel(const unsigned short* __restrict__ qbf,
                 const unsigned short* __restrict__ kbf,
                 const unsigned short* __restrict__ vbf,
                 const unsigned short* __restrict__ Ebf,
                 const float* __restrict__ amask,     // [B,1,1,S]
                 const float* __restrict__ X,         // hidden_states [B,S,D]
                 const float* __restrict__ gate_w,    // [H,DH]
                 const float* __restrict__ gate_b,    // [H]
                 float* __restrict__ out)             // [B,S,D]
{
  constexpr int SSTR = 516;                            // score row stride (f32)
  extern __shared__ unsigned char smem[];
  float*          sS = (float*)smem;                                  // 32 x 516 f32
  unsigned short* sQ = (unsigned short*)(smem + 32 * SSTR * 4);       // 32 x 72 bf16
  unsigned short* sK = sQ + 32 * 72;                                  // 16 x 72
  unsigned short* sV = sK + 16 * 72;                                  // 32 x 72
  float*          sW = (float*)(sV + 32 * 72);                        // 2 x 1328 f32
  float*       sGate = sW + 2 * 1328;                                 // 32 f32

  const int b = blockIdx.z, h = blockIdx.y;
  const int l0 = blockIdx.x * 32;
  const int t = threadIdx.x, wave = t >> 5, lane = t & 31;
  const int half = lane >> 4, ln = lane & 15;
  const int wr0 = wave * 16;

  float* wT1 = sW + wave * 1328;        // [16][33]  q·E window
  float* wT2 = wT1 + 528;               // [16][33]  k·E window

  const size_t headBase = (size_t)(b * H_N + h) * S_N;
  const v8f vzero = {0.f,0.f,0.f,0.f,0.f,0.f,0.f,0.f};

  // ---- stage q tile [32 x 64] bf16 (async copy to LDS) ----
  {
    const int row = t >> 1, cb = (t & 1) * 32;
    const unsigned short* src = qbf + (headBase + l0 + row) * DH_N + cb;
    #pragma unroll
    for (int j = 0; j < 4; ++j)
      async_copy16(sQ + row * 72 + cb + j * 8, src + j * 8);
  }
  async_wait();
  __syncthreads();

  // ---- q A-fragments, persistent across the whole r loop ----
  v16u Aq[2];
  #pragma unroll
  for (int c = 0; c < 2; ++c)
    #pragma unroll
    for (int v = 0; v < 8; ++v) {
      const int d = c * 32 + half * 16 + 2 * v;
      const unsigned short* p = sQ + (wr0 + ln) * 72 + d;
      Aq[c][2 * v]     = p[0];
      Aq[c][2 * v + 1] = p[1];
    }

  // ================= score pass =================
  for (int r0 = 0; r0 < S_N; r0 += 16) {
    __syncthreads();
    {   // stage k tile [16 x 64] (async copy to LDS) + prefetch next tile
      const int row = t >> 2, cb = (t & 3) * 16;
      const unsigned short* src = kbf + (headBase + r0 + row) * DH_N + cb;
      async_copy16(sK + row * 72 + cb,     src);
      async_copy16(sK + row * 72 + cb + 8, src + 8);
      if (r0 + 16 < S_N)
        __builtin_prefetch(src + 16 * DH_N, 0, 3);
    }
    async_wait();
    __syncthreads();

    const int jbase = l0 + wr0 - r0 + 496;   // E-window base, in [0, 992]

    v16u Bk[2], Ak[2], BE[2][2];
    #pragma unroll
    for (int c = 0; c < 2; ++c) {
      const unsigned short* pb = sK + ln * 72 + c * 32 + half * 16;   // B: N=ln
      v8u lo = *(const v8u*)pb;
      v8u hi = *(const v8u*)(pb + 8);
      #pragma unroll
      for (int i = 0; i < 8; ++i) { Bk[c][i] = lo[i]; Bk[c][i + 8] = hi[i]; }
      #pragma unroll
      for (int v = 0; v < 8; ++v) {                                    // A: M=ln
        const int d = c * 32 + half * 16 + 2 * v;
        const unsigned short* p = sK + ln * 72 + d;
        Ak[c][2 * v]     = p[0];
        Ak[c][2 * v + 1] = p[1];
      }
    }
    #pragma unroll
    for (int c = 0; c < 2; ++c)
      #pragma unroll
      for (int dk = 0; dk < 2; ++dk) {
        const unsigned short* p =
            Ebf + (size_t)(jbase + c * 16 + ln) * DH_N + dk * 32 + half * 16;
        v8u lo = *(const v8u*)p;
        v8u hi = *(const v8u*)(p + 8);
        #pragma unroll
        for (int i = 0; i < 8; ++i) { BE[c][dk][i] = lo[i]; BE[c][dk][i + 8] = hi[i]; }
      }

    v8f cQK = vzero;
    v8f cT1[2] = {vzero, vzero};
    v8f cT2[2] = {vzero, vzero};
    cQK = WMMA_B16(Aq[0], Bk[0], cQK);          // q . k^T  (K=64 via 2 steps)
    cQK = WMMA_B16(Aq[1], Bk[1], cQK);
    #pragma unroll
    for (int c = 0; c < 2; ++c) {
      cT1[c] = WMMA_B16(Aq[0], BE[c][0], cT1[c]);   // q . E_win^T
      cT1[c] = WMMA_B16(Aq[1], BE[c][1], cT1[c]);
      cT2[c] = WMMA_B16(Ak[0], BE[c][0], cT2[c]);   // k . E_win^T
      cT2[c] = WMMA_B16(Ak[1], BE[c][1], cT2[c]);
    }

    // spill T1/T2 to per-wave scratch so scores can gather along diagonals
    #pragma unroll
    for (int c = 0; c < 2; ++c)
      #pragma unroll
      for (int v = 0; v < 8; ++v) {
        const int m = half * 8 + v;
        wT1[m * 33 + c * 16 + ln] = cT1[c][v];
        wT2[m * 33 + c * 16 + ln] = cT2[c][v];
      }

    const float mval = amask[b * S_N + r0 + ln];
    #pragma unroll
    for (int v = 0; v < 8; ++v) {
      const int m  = half * 8 + v;          // local q row
      const int wj = m - ln + 15;           // window index = l - r + 15
      const float s = (cQK[v] + wT1[m * 33 + wj] + wT2[ln * 33 + wj]) * 0.125f + mval;
      sS[(wr0 + m) * SSTR + r0 + ln] = s;
    }
  }
  __syncthreads();

  // ================= clipped softmax (exact two-pass per row) =================
  {
    float* rp = sS + (wr0 + ln) * SSTR + half * 256;   // 2 lanes per row
    float mx = -3.0e38f;
    for (int c = 0; c < 256; ++c) mx = fmaxf(mx, rp[c]);
    mx = fmaxf(mx, __shfl_xor(mx, 16, 32));
    float z = 0.f;
    for (int c = 0; c < 256; ++c) {
      const float e = __expf(rp[c] - mx);
      rp[c] = e;
      z += e;
    }
    z += __shfl_xor(z, 16, 32);
    const float invz = 1.f / z;
    for (int c = 0; c < 256; ++c) {
      const float p = rp[c] * invz;
      rp[c] = fminf(fmaxf(fmaf(p, 1.0234375f, -0.0234375f), 0.f), 1.f);
    }
  }
  __syncthreads();

  // ================= per-token gate =================
  {
    const int row = wr0 + ln;
    const float* xp = X + (size_t)(b * S_N + l0 + row) * D_N + h * DH_N + half * 32;
    const float* gw = gate_w + h * DH_N + half * 32;
    float g = 0.f;
    #pragma unroll
    for (int d2 = 0; d2 < 32; ++d2) g = fmaf(xp[d2], gw[d2], g);
    g += __shfl_xor(g, 16, 32);
    if (half == 0) sGate[row] = 1.f / (1.f + __expf(-(g + gate_b[h])));
  }

  // ================= ctx = probs @ v =================
  v8f accC[4] = {vzero, vzero, vzero, vzero};
  for (int kc = 0; kc < S_N; kc += 32) {
    __syncthreads();
    {   // stage v chunk [32 x 64] (async copy to LDS) + prefetch next chunk
      const int row = t >> 1, cb = (t & 1) * 32;
      const unsigned short* src = vbf + (headBase + kc + row) * DH_N + cb;
      #pragma unroll
      for (int j = 0; j < 4; ++j)
        async_copy16(sV + row * 72 + cb + j * 8, src + j * 8);
      if (kc + 32 < S_N)
        __builtin_prefetch(src + 32 * DH_N, 0, 3);
    }
    async_wait();
    __syncthreads();

    v16u Ap;                                     // probs fragment (f32 -> bf16)
    #pragma unroll
    for (int v = 0; v < 8; ++v) {
      const int cix = kc + half * 16 + 2 * v;
      const float* p = sS + (wr0 + ln) * SSTR + cix;
      Ap[2 * v]     = f2bf(p[0]);
      Ap[2 * v + 1] = f2bf(p[1]);
    }
    #pragma unroll
    for (int nt = 0; nt < 4; ++nt) {
      v16u Bv;
      const int d = nt * 16 + ln;
      #pragma unroll
      for (int i = 0; i < 16; ++i)
        Bv[i] = sV[(half * 16 + i) * 72 + d];
      accC[nt] = WMMA_B16(Ap, Bv, accC[nt]);
    }
  }

  // ================= gate * ctx -> out [B,S,D] f32 =================
  #pragma unroll
  for (int nt = 0; nt < 4; ++nt)
    #pragma unroll
    for (int v = 0; v < 8; ++v) {
      const int m = half * 8 + v;
      const float g = sGate[wr0 + m];
      out[(size_t)(b * S_N + l0 + wr0 + m) * D_N + h * DH_N + nt * 16 + ln] =
          accC[nt][v] * g;
    }
}

// ---------------------------------------------------------------------------
extern "C" void kernel_launch(void* const* d_in, const int* in_sizes, int n_in,
                              void* d_out, int out_size, void* d_ws, size_t ws_size,
                              hipStream_t stream) {
  (void)in_sizes; (void)n_in; (void)out_size; (void)ws_size;
  const float* X    = (const float*)d_in[0];
  const float* amsk = (const float*)d_in[1];
  const float* Wq   = (const float*)d_in[2];
  const float* bq   = (const float*)d_in[3];
  const float* Wk   = (const float*)d_in[4];
  const float* bk   = (const float*)d_in[5];
  const float* Wv   = (const float*)d_in[6];
  const float* bv   = (const float*)d_in[7];
  const float* E    = (const float*)d_in[8];
  const float* gw   = (const float*)d_in[9];
  const float* gb   = (const float*)d_in[10];
  float* out = (float*)d_out;

  const size_t headElems = (size_t)B_N * H_N * S_N * DH_N;   // 8,388,608
  unsigned short* qbf = (unsigned short*)d_ws;
  unsigned short* kbf = qbf + headElems;
  unsigned short* vbf = kbf + headElems;
  unsigned short* Ebf = vbf + headElems;                     // 1024 x 64 bf16

  conv_pe_kernel<<<(1024 * DH_N + 255) / 256, 256, 0, stream>>>(E, Ebf);

  qkv_gemm_kernel<<<dim3(D_N / 128, (B_N * S_N) / 128, 3), 256, 0, stream>>>(
      X, Wq, bq, Wk, bk, Wv, bv, qbf, kbf, vbf);

  const size_t smem = (size_t)32 * 516 * 4 + (32 * 72 + 16 * 72 + 32 * 72) * 2 +
                      2 * 1328 * 4 + 32 * 4;                 // 88,320 bytes
  (void)hipFuncSetAttribute((const void*)attn_kernel,
                            hipFuncAttributeMaxDynamicSharedMemorySize, (int)smem);
  attn_kernel<<<dim3(S_N / 32, H_N, B_N), 64, smem, stream>>>(
      qbf, kbf, vbf, Ebf, amsk, X, gw, gb, out);
}
